// _SparseConv3dBase_22359599743102
// MI455X (gfx1250) — matrix operand order repaired and checked by reading, not compiled, not run on hardware
//
#include <hip/hip_runtime.h>
#include <hip/hip_bf16.h>

// ---------------------------------------------------------------------------
// Sparse 3x3x3 gather-GEMM for MI455X (gfx1250, wave32, WMMA).
//   out[j] = bias + sum_k feats[nbr_k(j)] @ W[k]
// fp32 inputs split into bf16 hi/lo; products via v_wmma_f32_16x16x32_bf16 as
// Ahi*Bhi + Alo*Bhi + Ahi*Blo (fp32 accumulate) -> ~16-bit mantissa fidelity.
// Software-pipelined double-buffered LDS staging; B staged with async-to-LDS
// when the toolchain exposes the gfx1250 builtin.
// ---------------------------------------------------------------------------

typedef __attribute__((ext_vector_type(16))) __bf16          v16bf;
typedef __attribute__((ext_vector_type(8)))  float           v8f;
typedef __attribute__((ext_vector_type(8)))  unsigned short  u16x8;
typedef __attribute__((ext_vector_type(16))) unsigned short  u16x16;

// Exact GCC-style vector type used by the async-to-LDS builtin prototype.
typedef int v4i_g __attribute__((vector_size(4 * sizeof(int))));

#define CIN   64
#define COUT  64
#define KTAP  27
#define MTILE 128                    // voxel rows per workgroup
#define WKELE (KTAP * CIN * COUT)    // 110592 weight elements

#if defined(__gfx1250__) && __has_builtin(__builtin_amdgcn_global_load_async_to_lds_b128) && __has_builtin(__builtin_amdgcn_s_wait_asynccnt)
#define USE_ASYNC_LDS 1
#else
#define USE_ASYNC_LDS 0
#endif

// Round-to-nearest-even fp32 -> bf16 (bits), plus bf16 residual.
__device__ __forceinline__ void split_bf16(float x, unsigned short& h, unsigned short& l) {
    unsigned u  = __builtin_bit_cast(unsigned, x);
    unsigned rb = u + 0x7FFFu + ((u >> 16) & 1u);
    unsigned short hb = (unsigned short)(rb >> 16);
    float hf = __builtin_bit_cast(float, (unsigned)hb << 16);
    float lo = x - hf;
    unsigned ul = __builtin_bit_cast(unsigned, lo);
    unsigned rl = ul + 0x7FFFu + ((ul >> 16) & 1u);
    h = hb;
    l = (unsigned short)(rl >> 16);
}

// Build a v16bf fragment from two 16-byte LDS segments.
__device__ __forceinline__ v16bf ld_frag2(const unsigned short* p0, const unsigned short* p1) {
    u16x8 a = *(const u16x8*)p0;
    u16x8 b = *(const u16x8*)p1;
    u16x16 q = __builtin_shufflevector(a, b, 0,1,2,3,4,5,6,7,8,9,10,11,12,13,14,15);
    return __builtin_bit_cast(v16bf, q);
}

#if USE_ASYNC_LDS
__device__ __forceinline__ void async_cp_b128(const unsigned short* g, unsigned short* l) {
    __builtin_amdgcn_global_load_async_to_lds_b128(
        (__attribute__((address_space(1))) v4i_g*)g,
        (__attribute__((address_space(3))) v4i_g*)l,
        0, 0);
}
#endif

// ---------------------------------------------------------------------------
// Prep: weight [k][cin][cout] fp32 -> transposed bf16 hi/lo planes
//       ws layout: [k][cout][cin] so B fragments are contiguous in K(=cin).
// ---------------------------------------------------------------------------
__global__ __launch_bounds__(256)
void sparseconv_prep_weights(const float* __restrict__ w,
                             unsigned short* __restrict__ wsHi,
                             unsigned short* __restrict__ wsLo) {
    int e = blockIdx.x * 256 + threadIdx.x;
    if (e >= WKELE) return;
    int k    = e >> 12;
    int r    = e & 4095;
    int cin  = r >> 6;
    int cout = r & 63;
    unsigned short h, l;
    split_bf16(w[e], h, l);
    int o = (k << 12) + (cout << 6) + cin;
    wsHi[o] = h;
    wsLo[o] = l;
}

// ---------------------------------------------------------------------------
// Main kernel: one workgroup = 8 wave32 = 128 output rows x 64 cols,
// double-buffered over the 27 taps.
// ---------------------------------------------------------------------------
__global__ __launch_bounds__(256)
void sparseconv_wmma_kernel(const float* __restrict__ feats,
                            const float* __restrict__ bias,
                            const int*   __restrict__ nbr,
                            const unsigned short* __restrict__ wsHi,
                            const unsigned short* __restrict__ wsLo,
                            float* __restrict__ out,
                            int Ntot) {
    __shared__ unsigned short sAh[2][MTILE * CIN];   // 2 x 16 KB
    __shared__ unsigned short sAl[2][MTILE * CIN];   // 2 x 16 KB
    __shared__ unsigned short sBh[2][CIN * COUT];    // 2 x  8 KB
    __shared__ unsigned short sBl[2][CIN * COUT];    // 2 x  8 KB

    const int t    = threadIdx.x;          // 0..255
    const int lane = t & 31;
    const int wave = t >> 5;               // 0..7
    const int base = blockIdx.x * MTILE;

    // A-fragment addressing (16x32 A, contiguous k order, same as B):
    //   lane<16 : K 0..15; lane>=16 : K 16..31
    const int m0    = wave * 16;
    const int arow  = m0 + (lane & 15);
    const int akoff = (lane < 16) ? 0 : 16;
    // B-fragment addressing (32x16 bf16 B): lane<16 -> K 0..15, lane>=16 -> 16..31
    const int bcolL = lane & 15;
    const int bkoff = (lane < 16) ? 0 : 16;

    // Gather assignment: 2 threads per row, 32 cin each.
    const int grow_r = t >> 1;             // 0..127
    const int c0     = (t & 1) * 32;
    const int gr     = base + grow_r;
    const bool rowok = gr < Ntot;

    // B staging assignment: 16 halves per thread per plane.
    const int boff = t * 16;

    v8f acc[4];
    #pragma unroll
    for (int n = 0; n < 4; ++n) acc[n] = (v8f)0.0f;

    float bv[4];
    #pragma unroll
    for (int n = 0; n < 4; ++n) bv[n] = bias[n * 16 + bcolL];

    // ---- staging helpers -------------------------------------------------
    float4 ra[8];
    auto gatherA = [&](int idx) {
        const float4* src = (idx >= 0)
            ? (const float4*)(feats + (size_t)idx * CIN + c0) : nullptr;
        #pragma unroll
        for (int j = 0; j < 8; ++j)
            ra[j] = src ? src[j] : float4{0.f, 0.f, 0.f, 0.f};
    };
    auto commitA = [&](int nb) {
        #pragma unroll
        for (int j = 0; j < 8; ++j) {
            const int o = grow_r * CIN + c0 + 4 * j;
            unsigned short h, l;
            split_bf16(ra[j].x, h, l); sAh[nb][o + 0] = h; sAl[nb][o + 0] = l;
            split_bf16(ra[j].y, h, l); sAh[nb][o + 1] = h; sAl[nb][o + 1] = l;
            split_bf16(ra[j].z, h, l); sAh[nb][o + 2] = h; sAl[nb][o + 2] = l;
            split_bf16(ra[j].w, h, l); sAh[nb][o + 3] = h; sAl[nb][o + 3] = l;
        }
    };

#if USE_ASYNC_LDS
    auto stageB = [&](int k, int nb) {
        const unsigned short* gh = wsHi + (k << 12) + boff;
        const unsigned short* gl = wsLo + (k << 12) + boff;
        async_cp_b128(gh,     &sBh[nb][boff]);
        async_cp_b128(gh + 8, &sBh[nb][boff + 8]);
        async_cp_b128(gl,     &sBl[nb][boff]);
        async_cp_b128(gl + 8, &sBl[nb][boff + 8]);
    };
#else
    u16x8 rb[4];
    auto issueB = [&](int k) {
        const unsigned short* gh = wsHi + (k << 12) + boff;
        const unsigned short* gl = wsLo + (k << 12) + boff;
        rb[0] = *(const u16x8*)&gh[0];
        rb[1] = *(const u16x8*)&gh[8];
        rb[2] = *(const u16x8*)&gl[0];
        rb[3] = *(const u16x8*)&gl[8];
    };
    auto commitB = [&](int nb) {
        *(u16x8*)&sBh[nb][boff]     = rb[0];
        *(u16x8*)&sBh[nb][boff + 8] = rb[1];
        *(u16x8*)&sBl[nb][boff]     = rb[2];
        *(u16x8*)&sBl[nb][boff + 8] = rb[3];
    };
#endif

    auto compute = [&](int pb) {
        #pragma unroll
        for (int c = 0; c < 2; ++c) {
            const int a0 = arow * CIN + c * 32 + akoff;
            v16bf aHi = ld_frag2(&sAh[pb][a0], &sAh[pb][a0 + 8]);
            v16bf aLo = ld_frag2(&sAl[pb][a0], &sAl[pb][a0 + 8]);
            #pragma unroll
            for (int n = 0; n < 4; ++n) {
                const int b0 = (n * 16 + bcolL) * CIN + c * 32 + bkoff;
                v16bf bHi = ld_frag2(&sBh[pb][b0], &sBh[pb][b0 + 8]);
                v16bf bLo = ld_frag2(&sBl[pb][b0], &sBl[pb][b0 + 8]);
                acc[n] = __builtin_amdgcn_wmma_f32_16x16x32_bf16(
                    false, aHi, false, bHi, (short)0, acc[n], false, false);
                acc[n] = __builtin_amdgcn_wmma_f32_16x16x32_bf16(
                    false, aLo, false, bHi, (short)0, acc[n], false, false);
                acc[n] = __builtin_amdgcn_wmma_f32_16x16x32_bf16(
                    false, aHi, false, bLo, (short)0, acc[n], false, false);
            }
        }
    };

    // ---- prologue: stage tap 0 into buffer 0 -----------------------------
    int idx_nxt = rowok ? nbr[gr] : -1;                 // tap 0
    gatherA(idx_nxt);
#if USE_ASYNC_LDS
    stageB(0, 0);
#else
    issueB(0);
#endif
    idx_nxt = rowok ? nbr[(size_t)Ntot + gr] : -1;      // tap 1
    commitA(0);
#if USE_ASYNC_LDS
    __builtin_amdgcn_s_wait_asynccnt(0);
#else
    commitB(0);
#endif
    __syncthreads();

    // ---- pipelined main loop over taps -----------------------------------
    for (int k = 0; k < KTAP; ++k) {
        const int  pb   = k & 1;
        const int  nb   = pb ^ 1;
        const bool more = (k + 1) < KTAP;   // uniform across block

        if (more) {
            gatherA(idx_nxt);               // issue next tap's gathers
#if USE_ASYNC_LDS
            stageB(k + 1, nb);              // async copy straight into LDS
#else
            issueB(k + 1);
#endif
        }
        const int idx_fut = (more && (k + 2) < KTAP && rowok)
            ? nbr[(size_t)(k + 2) * Ntot + gr] : -1;

        compute(pb);                        // 24 WMMAs hide the loads above

        if (more) {
            commitA(nb);                    // waits gathers, split, LDS store
#if !USE_ASYNC_LDS
            commitB(nb);
#endif
            idx_nxt = idx_fut;
        }
#if USE_ASYNC_LDS
        if (more) __builtin_amdgcn_s_wait_asynccnt(0);
#endif
        __syncthreads();
    }

    // ---- epilogue: C layout = lane(0..15) cols, VGPR v -> row (+8 hi lanes)
    const int rbase = base + m0 + ((lane >> 4) << 3);
    #pragma unroll
    for (int n = 0; n < 4; ++n) {
        const int col = n * 16 + bcolL;
        #pragma unroll
        for (int v = 0; v < 8; ++v) {
            int g = rbase + v;
            if (g < Ntot) out[(size_t)g * COUT + col] = acc[n][v] + bv[n];
        }
    }
}

extern "C" void kernel_launch(void* const* d_in, const int* in_sizes, int n_in,
                              void* d_out, int out_size, void* d_ws, size_t ws_size,
                              hipStream_t stream) {
    (void)n_in; (void)out_size; (void)ws_size;
    const float* feats  = (const float*)d_in[0];
    const float* weight = (const float*)d_in[1];
    const float* bias   = (const float*)d_in[2];
    const int*   nbr    = (const int*)d_in[3];
    float*       out    = (float*)d_out;

    const int Ntot = in_sizes[0] / CIN;

    unsigned short* wsHi = (unsigned short*)d_ws;
    unsigned short* wsLo = wsHi + WKELE;

    // 1) convert + transpose weights to bf16 hi/lo planes in scratch
    sparseconv_prep_weights<<<(WKELE + 255) / 256, 256, 0, stream>>>(weight, wsHi, wsLo);

    // 2) pipelined gather-GEMM with WMMA
    const int blocks = (Ntot + MTILE - 1) / MTILE;
    sparseconv_wmma_kernel<<<blocks, 256, 0, stream>>>(
        feats, bias, nbr, wsHi, wsLo, out, Ntot);
}
